// LRU_9045201125893
// MI455X (gfx1250) — matrix operand order repaired and checked by reading, not compiled, not run on hardware
//
#include <hip/hip_runtime.h>

typedef unsigned short u16;
typedef __attribute__((ext_vector_type(16))) _Float16 v16h;
typedef __attribute__((ext_vector_type(8)))  float    v8f;
typedef int v4i __attribute__((__vector_size__(16)));

#define SEQ_LEN 16384
#define D_MODEL 1024
#define D_STATE 512
#define CHUNK   128
#define NCHUNK  128   // SEQ_LEN / CHUNK

#if defined(__has_builtin)
#if __has_builtin(__builtin_amdgcn_global_load_async_to_lds_b128)
#define USE_ASYNC_LDS 1
#endif
#endif

// ---------- helpers ----------
__device__ __forceinline__ u16 f2h(float f) {
    _Float16 h = (_Float16)f;
    return __builtin_bit_cast(u16, h);
}

__device__ __forceinline__ void wait_async0() {
#if defined(__has_builtin)
#if __has_builtin(__builtin_amdgcn_s_wait_asynccnt)
    __builtin_amdgcn_s_wait_asynccnt(0);
    return;
#else
    asm volatile("s_wait_asynccnt 0x0" ::: "memory");
    return;
#endif
#else
    asm volatile("s_wait_asynccnt 0x0" ::: "memory");
#endif
}

union FragH  { uint4 u[2]; v16h v; };
union Pack8  { u16 h[8]; uint4 u; };

// ---------- parameter / input prep ----------
__global__ __launch_bounds__(256)
void lru_prep_x(const float* __restrict__ X, u16* __restrict__ Xh) {
    size_t i = ((size_t)blockIdx.x * 256 + threadIdx.x) * 8;
    float4 a = *(const float4*)(X + i);
    float4 b = *(const float4*)(X + i + 4);
    Pack8 p;
    p.h[0] = f2h(a.x); p.h[1] = f2h(a.y); p.h[2] = f2h(a.z); p.h[3] = f2h(a.w);
    p.h[4] = f2h(b.x); p.h[5] = f2h(b.y); p.h[6] = f2h(b.z); p.h[7] = f2h(b.w);
    *(uint4*)(Xh + i) = p.u;
}

// W1[n][k]: rows 0..511 = B_re[n]*exp(gamma[n]); rows 512..1023 = B_im * exp(gamma)
__global__ void lru_prep_w1(const float* __restrict__ Bre, const float* __restrict__ Bim,
                            const float* __restrict__ gam, u16* __restrict__ W1) {
    int n = blockIdx.x;
    float g = __expf(gam[n]);
    for (int k = threadIdx.x; k < D_MODEL; k += 256) {
        W1[(size_t)n * D_MODEL + k]             = f2h(Bre[(size_t)n * D_MODEL + k] * g);
        W1[(size_t)(n + D_STATE) * D_MODEL + k] = f2h(Bim[(size_t)n * D_MODEL + k] * g);
    }
}

// W2[m][s]: s<512: C_re[m][s]; s>=512: -C_im[m][s-512]
__global__ void lru_prep_w2(const float* __restrict__ Cre, const float* __restrict__ Cim,
                            u16* __restrict__ W2) {
    int m = blockIdx.x;
    for (int s = threadIdx.x; s < D_STATE; s += 256) {
        W2[(size_t)m * D_MODEL + s]           = f2h(Cre[(size_t)m * D_STATE + s]);
        W2[(size_t)m * D_MODEL + D_STATE + s] = f2h(-Cim[(size_t)m * D_STATE + s]);
    }
}

__global__ void lru_prep_lambda(const float* __restrict__ nu, const float* __restrict__ th,
                                float* __restrict__ lre, float* __restrict__ lim) {
    int s = threadIdx.x;                      // block of 512
    float mag = __expf(-__expf(nu[s]));
    float ang = __expf(th[s]);
    lre[s] = mag * __cosf(ang);
    lim[s] = mag * __sinf(ang);
}

// ---------- tiled f16 WMMA GEMM ----------
// Out[l,n] = sum_k A[l,k] * W[n,k] ; A: L x 1024 f16 ; W: 1024 x 1024 f16 ; Out f32
// FUSE: Out += D[n] * X32[l,n]
template <bool FUSE>
__global__ __launch_bounds__(256)
void lru_gemm(const u16* __restrict__ A, const u16* __restrict__ W,
              float* __restrict__ Out, const float* __restrict__ Dv,
              const float* __restrict__ X32) {
    constexpr int BM = 128, BN = 128, BK = 32;
    constexpr int LDH = BK + 8;                         // padded LDS stride (halves)
    constexpr int NT = D_MODEL / BK;

    const int tid = threadIdx.x;
    const int m0 = blockIdx.y * BM, n0 = blockIdx.x * BN;
    const int w = tid >> 5, l = tid & 31;
    const int wm = w & 3, wn = w >> 2;                  // 4x2 wave grid; wave = 32 rows x 64 cols
    const int lrow = l & 15, lhal = l >> 4;

    v8f acc[2][4] = {};

    // staging: 2 threads per row, 16 halves (2 x 16B) each
    const int sr = tid >> 1, sc = (tid & 1) * 16;
    const u16* agp = A + (size_t)(m0 + sr) * D_MODEL + sc;
    const u16* bgp = W + (size_t)(n0 + sr) * D_MODEL + sc;

#if USE_ASYNC_LDS
    __shared__ __align__(16) u16 sA[2][BM * LDH];
    __shared__ __align__(16) u16 sB[2][BN * LDH];

    typedef __attribute__((address_space(1))) v4i as1_v4i;
    typedef __attribute__((address_space(3))) v4i as3_v4i;

    auto issue_tile = [&](int buf, int k0) {
        __builtin_amdgcn_global_load_async_to_lds_b128(
            (as1_v4i*)(agp + k0), (as3_v4i*)&sA[buf][sr * LDH + sc], 0, 0);
        __builtin_amdgcn_global_load_async_to_lds_b128(
            (as1_v4i*)(agp + k0), (as3_v4i*)&sA[buf][sr * LDH + sc], 16, 0);
        __builtin_amdgcn_global_load_async_to_lds_b128(
            (as1_v4i*)(bgp + k0), (as3_v4i*)&sB[buf][sr * LDH + sc], 0, 0);
        __builtin_amdgcn_global_load_async_to_lds_b128(
            (as1_v4i*)(bgp + k0), (as3_v4i*)&sB[buf][sr * LDH + sc], 16, 0);
    };

    issue_tile(0, 0);
    for (int kt = 0; kt < NT; ++kt) {
        const int cur = kt & 1;
        wait_async0();                                  // this wave's DMA for buf `cur` done
        __syncthreads();                                // all waves' DMA visible; prev compute done
        if (kt + 1 < NT) issue_tile(cur ^ 1, (kt + 1) * BK);

        const u16* cA = &sA[cur][0];
        const u16* cB = &sB[cur][0];
#else
    __shared__ __align__(16) u16 sAb[BM * LDH];
    __shared__ __align__(16) u16 sBb[BN * LDH];
    const uint4* ag = (const uint4*)agp;
    const uint4* bg = (const uint4*)bgp;
    uint4 ra0 = ag[0], ra1 = ag[1];
    uint4 rb0 = bg[0], rb1 = bg[1];

    for (int kt = 0; kt < NT; ++kt) {
        __syncthreads();
        *(uint4*)(sAb + sr * LDH + sc)     = ra0;
        *(uint4*)(sAb + sr * LDH + sc + 8) = ra1;
        *(uint4*)(sBb + sr * LDH + sc)     = rb0;
        *(uint4*)(sBb + sr * LDH + sc + 8) = rb1;
        __syncthreads();
        if (kt + 1 < NT) {                              // prefetch next tile
            ag += 4; bg += 4;
            ra0 = ag[0]; ra1 = ag[1];
            rb0 = bg[0]; rb1 = bg[1];
        }
        const u16* cA = &sAb[0];
        const u16* cB = &sBb[0];
#endif
        // A frags: lane<16 -> M=lane, K:[0..7]+[16..23]; lane>=16 -> K:[8..15]+[24..31]
        FragH af[2];
        const int klo = lhal * 8;
#pragma unroll
        for (int i = 0; i < 2; ++i) {
            int row = wm * 32 + i * 16 + lrow;
            af[i].u[0] = *(const uint4*)(cA + row * LDH + klo);
            af[i].u[1] = *(const uint4*)(cA + row * LDH + klo + 16);
        }
        // B frags: lane<16 -> N=lane, K:[0..7]+[16..23]; lane>=16 -> K:[8..15]+[24..31]
        FragH bq[4];
        const int kb = lhal * 8;
#pragma unroll
        for (int j = 0; j < 4; ++j) {
            int col = wn * 64 + j * 16 + lrow;
            bq[j].u[0] = *(const uint4*)(cB + col * LDH + kb);
            bq[j].u[1] = *(const uint4*)(cB + col * LDH + kb + 16);
        }
#pragma unroll
        for (int i = 0; i < 2; ++i)
#pragma unroll
            for (int j = 0; j < 4; ++j)
                acc[i][j] = __builtin_amdgcn_wmma_f32_16x16x32_f16(
                    false, af[i].v, false, bq[j].v, (short)0, acc[i][j], false, false);
    }

    // epilogue: C layout — lane l, vgpr r: row = r + 8*(l>>4), col = l&15
#pragma unroll
    for (int i = 0; i < 2; ++i)
#pragma unroll
        for (int j = 0; j < 4; ++j) {
            int rowb = m0 + wm * 32 + i * 16 + 8 * lhal;
            int colb = n0 + wn * 64 + j * 16 + lrow;
            float dval = FUSE ? Dv[colb] : 0.0f;
#pragma unroll
            for (int r = 0; r < 8; ++r) {
                int row = rowb + r;
                float v = acc[i][j][r];
                if (FUSE) v += dval * X32[(size_t)row * D_MODEL + colb];
                Out[(size_t)row * D_MODEL + colb] = v;
            }
        }
}

// ---------- chunked complex scan (3 phases, all f32) ----------
__global__ __launch_bounds__(256)
void lru_scan_agg(const float* __restrict__ Bu, const float* __restrict__ lre,
                  const float* __restrict__ lim, float* __restrict__ aggr,
                  float* __restrict__ aggi) {
    int s = blockIdx.y * 256 + threadIdx.x;
    int c = blockIdx.x;
    float lr = lre[s], li = lim[s], hr = 0.f, hi = 0.f;
    const float* base = Bu + (size_t)c * CHUNK * D_MODEL;
    for (int t = 0; t < CHUNK; ++t) {
        float br = base[(size_t)t * D_MODEL + s];
        float bi = base[(size_t)t * D_MODEL + D_STATE + s];
        float nr = lr * hr - li * hi + br;
        float ni = lr * hi + li * hr + bi;
        hr = nr; hi = ni;
    }
    aggr[(size_t)c * D_STATE + s] = hr;
    aggi[(size_t)c * D_STATE + s] = hi;
}

__global__ void lru_scan_carry(const float* __restrict__ lre, const float* __restrict__ lim,
                               const float* __restrict__ aggr, const float* __restrict__ aggi,
                               float* __restrict__ carr, float* __restrict__ cari) {
    int s = blockIdx.x * 256 + threadIdx.x;
    float pr = lre[s], pi = lim[s];
#pragma unroll
    for (int i = 0; i < 7; ++i) {                       // lambda^128 by squaring
        float nr = pr * pr - pi * pi, ni = 2.f * pr * pi;
        pr = nr; pi = ni;
    }
    float cr = 0.f, ci = 0.f;
    for (int c = 0; c < NCHUNK; ++c) {
        carr[(size_t)c * D_STATE + s] = cr;
        cari[(size_t)c * D_STATE + s] = ci;
        float nr = pr * cr - pi * ci + aggr[(size_t)c * D_STATE + s];
        float ni = pr * ci + pi * cr + aggi[(size_t)c * D_STATE + s];
        cr = nr; ci = ni;
    }
}

__global__ __launch_bounds__(256)
void lru_scan_final(const float* __restrict__ Bu, const float* __restrict__ lre,
                    const float* __restrict__ lim, const float* __restrict__ carr,
                    const float* __restrict__ cari, u16* __restrict__ H) {
    int s = blockIdx.y * 256 + threadIdx.x;
    int c = blockIdx.x;
    float lr = lre[s], li = lim[s];
    float hr = carr[(size_t)c * D_STATE + s];
    float hi = cari[(size_t)c * D_STATE + s];
    const float* base = Bu + (size_t)c * CHUNK * D_MODEL;
    u16* hb = H + (size_t)c * CHUNK * D_MODEL;
    for (int t = 0; t < CHUNK; ++t) {
        float br = base[(size_t)t * D_MODEL + s];
        float bi = base[(size_t)t * D_MODEL + D_STATE + s];
        float nr = lr * hr - li * hi + br;
        float ni = lr * hi + li * hr + bi;
        hr = nr; hi = ni;
        hb[(size_t)t * D_MODEL + s]           = f2h(hr);
        hb[(size_t)t * D_MODEL + D_STATE + s] = f2h(hi);
    }
}

// ---------- launch ----------
extern "C" void kernel_launch(void* const* d_in, const int* in_sizes, int n_in,
                              void* d_out, int out_size, void* d_ws, size_t ws_size,
                              hipStream_t stream) {
    const float* x    = (const float*)d_in[0];
    const float* nu   = (const float*)d_in[1];
    const float* th   = (const float*)d_in[2];
    const float* gam  = (const float*)d_in[3];
    const float* Bre  = (const float*)d_in[4];
    const float* Bim  = (const float*)d_in[5];
    const float* Cre  = (const float*)d_in[6];
    const float* Cim  = (const float*)d_in[7];
    const float* Dv   = (const float*)d_in[8];
    float* out = (float*)d_out;

    char* ws = (char*)d_ws;
    const size_t MB = 1024ull * 1024ull;
    u16*   W1   = (u16*)(ws + 0);                    // 2 MB
    u16*   W2   = (u16*)(ws + 2 * MB);               // 2 MB
    float* lre  = (float*)(ws + 4 * MB);             // 2 KB
    float* lim  = (float*)(ws + 4 * MB + 4096);
    float* aggr = (float*)(ws + 5 * MB);             // 256 KB
    float* aggi = (float*)(ws + 5 * MB + 256 * 1024);
    float* carr = (float*)(ws + 6 * MB);             // 256 KB
    float* cari = (float*)(ws + 6 * MB + 256 * 1024);
    float* Bu   = (float*)(ws + 8 * MB);             // 64 MB
    u16*   XH   = (u16*)(ws + 72 * MB);              // 32 MB: x-f16 for GEMM1, then reused as H
    u16*   H    = XH;                                // scan_final overwrites after GEMM1 done

    lru_prep_lambda<<<1, D_STATE, 0, stream>>>(nu, th, lre, lim);
    lru_prep_w1<<<D_STATE, 256, 0, stream>>>(Bre, Bim, gam, W1);
    lru_prep_w2<<<D_MODEL, 256, 0, stream>>>(Cre, Cim, W2);
    lru_prep_x<<<(SEQ_LEN * D_MODEL) / (256 * 8), 256, 0, stream>>>(x, XH);

    dim3 gemmGrid(D_MODEL / 128, SEQ_LEN / 128);
    lru_gemm<false><<<gemmGrid, 256, 0, stream>>>(XH, W1, Bu, nullptr, nullptr);

    lru_scan_agg<<<dim3(NCHUNK, 2), 256, 0, stream>>>(Bu, lre, lim, aggr, aggi);
    lru_scan_carry<<<2, 256, 0, stream>>>(lre, lim, aggr, aggi, carr, cari);
    lru_scan_final<<<dim3(NCHUNK, 2), 256, 0, stream>>>(Bu, lre, lim, carr, cari, H);

    lru_gemm<true><<<gemmGrid, 256, 0, stream>>>(H, W2, out, Dv, x);
}
